// WanAttentionBlock_22943715295486
// MI455X (gfx1250) — matrix operand order repaired and checked by reading, not compiled, not run on hardware
//
#include <hip/hip_runtime.h>

#define B_     2
#define S_     2048
#define CL_    512
#define DIM_   2048
#define HEADS_ 16
#define FFN_   8192
#define HD_    128
#define NTOK_  (B_ * S_)
#define NCTX_  (B_ * CL_)
#define EPS_   1e-6f

typedef __attribute__((ext_vector_type(16))) __bf16 v16bf;
typedef __attribute__((ext_vector_type(8)))  __bf16 v8bf;
typedef __attribute__((ext_vector_type(2)))  __bf16 v2bf;
typedef __attribute__((ext_vector_type(8)))  float  v8f;
typedef __attribute__((ext_vector_type(4)))  int    v4i;

// ---- CDNA5 async global->LDS copy (ASYNCcnt) with compile-safe fallback ----
#if defined(__AMDGCN__) && __has_builtin(__builtin_amdgcn_global_load_async_to_lds_b128)
#define USE_ASYNC_LDS 1
#else
#define USE_ASYNC_LDS 0
#endif

#if USE_ASYNC_LDS
#if __has_builtin(__builtin_amdgcn_s_wait_asynccnt)
#define WAIT_ASYNC(n) __builtin_amdgcn_s_wait_asynccnt(n)
#else
#define WAIT_ASYNC(n) asm volatile("s_wait_asynccnt " #n ::: "memory")
#endif
#else
#define WAIT_ASYNC(n) ((void)0)
#endif

typedef __attribute__((address_space(1))) v4i* as1_v4i;
typedef __attribute__((address_space(3))) v4i* as3_v4i;

static __device__ __forceinline__ void async_copy16(const void* g, void* l) {
#if USE_ASYNC_LDS
  as1_v4i gp = (as1_v4i)(unsigned long long)g;
  as3_v4i lp = (as3_v4i)(unsigned int)(unsigned long long)l;
  __builtin_amdgcn_global_load_async_to_lds_b128(gp, lp, 0, 0);
#else
  *(v8bf*)l = *(const v8bf*)g;
#endif
}

static __device__ __forceinline__ v16bf concat16(v8bf lo, v8bf hi) {
  return __builtin_shufflevector(lo, hi, 0,1,2,3,4,5,6,7,8,9,10,11,12,13,14,15);
}

static __device__ __forceinline__ float gelu_tanh(float v) {
  float c = 0.7978845608028654f * (v + 0.044715f * v * v * v);
  return 0.5f * v * (1.0f + tanhf(c));
}

// ---------------------------------------------------------------------------
// Weight convert+transpose: W[K][N] f32 (row-major) -> Wt[N][K] bf16
// ---------------------------------------------------------------------------
__global__ __launch_bounds__(256)
void wtrans_kernel(const float* __restrict__ W, __bf16* __restrict__ Wt,
                   int K, int N) {
  __shared__ float tile[32][33];
  int k0 = blockIdx.x * 32;
  int n0 = blockIdx.y * 32;
  int tx = threadIdx.x & 31, ty = threadIdx.x >> 5;  // 32 x 8
  #pragma unroll
  for (int i = 0; i < 32; i += 8)
    tile[ty + i][tx] = W[(size_t)(k0 + ty + i) * N + n0 + tx];
  __syncthreads();
  #pragma unroll
  for (int i = 0; i < 32; i += 8)
    Wt[(size_t)(n0 + ty + i) * K + k0 + tx] = (__bf16)tile[tx][ty + i];
}

// ---------------------------------------------------------------------------
// mod = modulation (broadcast over batch) + e   -> [B,6,DIM] f32
// ---------------------------------------------------------------------------
__global__ __launch_bounds__(256)
void mod_kernel(const float* __restrict__ modulation, const float* __restrict__ e,
                float* __restrict__ mod, int n) {
  int i = blockIdx.x * 256 + threadIdx.x;
  if (i < n) mod[i] = modulation[i % (6 * DIM_)] + e[i];
}

// ---------------------------------------------------------------------------
// f32 -> bf16 copy
// ---------------------------------------------------------------------------
__global__ __launch_bounds__(256)
void f2b_kernel(const float* __restrict__ x, __bf16* __restrict__ y, size_t n) {
  size_t i = (size_t)blockIdx.x * 256 + threadIdx.x;
  if (i < n) y[i] = (__bf16)x[i];
}

// ---------------------------------------------------------------------------
// Fused RMSNorm / AdaLN-modulate / optional second RMSNorm -> bf16
// One block (256 thr) per row of DIM=2048.
// ---------------------------------------------------------------------------
__global__ __launch_bounds__(256)
void rmsmod_kernel(const float* __restrict__ x, const float* __restrict__ w,
                   const float* __restrict__ mod, int esc, int esh,
                   const float* __restrict__ w2, float scale,
                   __bf16* __restrict__ out, int rows_per_b) {
  __shared__ float red[256];
  int row = blockIdx.x;
  int tid = threadIdx.x;
  const float* xr = x + (size_t)row * DIM_;
  int base = tid * 8;
  float v[8];
  float ss = 0.f;
  #pragma unroll
  for (int i = 0; i < 8; i++) { v[i] = xr[base + i]; ss += v[i] * v[i]; }
  red[tid] = ss; __syncthreads();
  for (int s = 128; s > 0; s >>= 1) {
    if (tid < s) red[tid] += red[tid + s];
    __syncthreads();
  }
  float r1 = rsqrtf(red[0] * (1.0f / DIM_) + EPS_);
  __syncthreads();
  const float* mrow = mod ? (mod + (size_t)(row / rows_per_b) * 6 * DIM_) : nullptr;
  #pragma unroll
  for (int i = 0; i < 8; i++) {
    int c = base + i;
    float t = v[i] * r1 * w[c];
    if (mrow) t = t * (1.0f + mrow[esc * DIM_ + c]) + mrow[esh * DIM_ + c];
    v[i] = t;
  }
  if (w2) {
    float ss2 = 0.f;
    #pragma unroll
    for (int i = 0; i < 8; i++) ss2 += v[i] * v[i];
    red[tid] = ss2; __syncthreads();
    for (int s = 128; s > 0; s >>= 1) {
      if (tid < s) red[tid] += red[tid + s];
      __syncthreads();
    }
    float r2 = rsqrtf(red[0] * (1.0f / DIM_) + EPS_);
    __syncthreads();
    #pragma unroll
    for (int i = 0; i < 8; i++) v[i] = v[i] * r2 * w2[base + i];
  }
  __bf16* orow = out + (size_t)row * DIM_;
  #pragma unroll
  for (int i = 0; i < 8; i++) orow[base + i] = (__bf16)(v[i] * scale);
}

// ---------------------------------------------------------------------------
// Tiled bf16 WMMA GEMM: D = A[M,K] @ Wt[N,K]^T (+bias) -> epilogues
//   block tile 128x128, K-step 32, 8 waves (4 M x 2 N), wave tile 32x64.
//   Double-buffered LDS staged with async global->LDS (one barrier / K-step).
// ---------------------------------------------------------------------------
__global__ __launch_bounds__(256)
void gemm_bf16_kernel(const __bf16* __restrict__ A, const __bf16* __restrict__ Wt,
                      const float* __restrict__ bias,
                      const float* __restrict__ res, const float* __restrict__ gate,
                      float* __restrict__ outf, __bf16* __restrict__ outb,
                      int M, int N, int K, int rows_per_b, int gstride,
                      int act, float scale) {
  __shared__ __align__(16) __bf16 As[2][128 * 32];
  __shared__ __align__(16) __bf16 Bs[2][128 * 32];
  int tid = threadIdx.x;
  int lane = tid & 31, wave = tid >> 5;
  int mw = wave & 3, nw = wave >> 2;           // 4 x 2 wave grid
  int lm = lane & 15, h = lane >> 4;
  int bm = blockIdx.x * 128, bn = blockIdx.y * 128;

  v8f acc[2][4] = {};

  int lr = tid >> 1;                 // 0..127
  int lc = (tid & 1) << 4;           // 0 / 16

  auto stage = [&](int k0, int bb) {
    const __bf16* asrc = A  + (size_t)(bm + lr) * K + k0 + lc;
    const __bf16* bsrc = Wt + (size_t)(bn + lr) * K + k0 + lc;
    async_copy16(asrc,     &As[bb][lr * 32 + lc]);
    async_copy16(asrc + 8, &As[bb][lr * 32 + lc + 8]);
    async_copy16(bsrc,     &Bs[bb][lr * 32 + lc]);
    async_copy16(bsrc + 8, &Bs[bb][lr * 32 + lc + 8]);
  };

  stage(0, 0);
  int buf = 0;
  for (int k0 = 0; k0 < K; k0 += 32) {
    if (k0 + 32 < K) {
      stage(k0 + 32, buf ^ 1);
      WAIT_ASYNC(4);     // async loads complete in order: previous stage done
    } else {
      WAIT_ASYNC(0);
    }
    __syncthreads();

    v16bf af[2], bfr[4];
    #pragma unroll
    for (int i = 0; i < 2; i++) {
      const __bf16* base = &As[buf][(mw * 32 + i * 16 + lm) * 32];
      v8bf lo = *(const v8bf*)(base + 8 * h);
      v8bf hi = *(const v8bf*)(base + 16 + 8 * h);
      af[i] = concat16(lo, hi);
    }
    #pragma unroll
    for (int j = 0; j < 4; j++) {
      const __bf16* base = &Bs[buf][(nw * 64 + j * 16 + lm) * 32];
      v8bf lo = *(const v8bf*)(base + 8 * h);
      v8bf hi = *(const v8bf*)(base + 16 + 8 * h);
      bfr[j] = concat16(lo, hi);
    }
    #pragma unroll
    for (int i = 0; i < 2; i++)
      #pragma unroll
      for (int j = 0; j < 4; j++)
        acc[i][j] = __builtin_amdgcn_wmma_f32_16x16x32_bf16(
            false, af[i], false, bfr[j], (short)0, acc[i][j], false, false);
    buf ^= 1;
  }

  #pragma unroll
  for (int i = 0; i < 2; i++) {
    #pragma unroll
    for (int j = 0; j < 4; j++) {
      #pragma unroll
      for (int r = 0; r < 8; r++) {
        int m = bm + mw * 32 + i * 16 + r + 8 * h;
        int n = bn + nw * 64 + j * 16 + lm;
        float v = acc[i][j][r];
        if (bias) v += bias[n];
        if (act == 1) v = gelu_tanh(v);
        v *= scale;
        size_t idx = (size_t)m * N + n;
        if (outb) outb[idx] = (__bf16)v;
        if (outf) {
          float o = v;
          if (gate) o *= gate[(size_t)(m / rows_per_b) * gstride + n];
          if (res) o += res[idx];
          outf[idx] = o;
        }
      }
    }
  }
}

// ---------------------------------------------------------------------------
// Flash attention (online softmax), bf16 in / bf16 out, f32 accum.
// grid: (Sq/64, HEADS, B), 128 threads (4 waves), each wave owns 16 q rows.
// 1/sqrt(hd) is pre-folded into Q.
// ---------------------------------------------------------------------------
__global__ __launch_bounds__(128)
void flash_attn_kernel(const __bf16* __restrict__ Q, const __bf16* __restrict__ K,
                       const __bf16* __restrict__ V, __bf16* __restrict__ O,
                       int Sq, int Sk) {
  __shared__ __align__(16) __bf16 Qs[64 * 128];
  __shared__ __align__(16) __bf16 Ks[64 * 128];
  __shared__ __align__(16) __bf16 Vs[128 * 64];   // transposed: [d][kn]

  int b = blockIdx.z, head = blockIdx.y;
  int q0 = blockIdx.x * 64;
  int tid = threadIdx.x, lane = tid & 31, wave = tid >> 5;
  int lm = lane & 15, h = lane >> 4;
  size_t hoff = (size_t)head * HD_;

  {   // stage Q tile [64][128]
    int r = tid & 63, hf = tid >> 6;
    const __bf16* src = Q + ((size_t)b * Sq + q0 + r) * DIM_ + hoff + hf * 64;
    #pragma unroll
    for (int i = 0; i < 8; i++)
      *(v8bf*)&Qs[r * 128 + hf * 64 + i * 8] = *(const v8bf*)(src + i * 8);
  }

  float m_prev = -3.0e38f, l = 0.f;
  v8f Oacc[8] = {};

  for (int k0 = 0; k0 < Sk; k0 += 64) {
    __syncthreads();
    {   // stage K tile [64][128]
      int r = tid & 63, hf = tid >> 6;
      const __bf16* src = K + ((size_t)b * Sk + k0 + r) * DIM_ + hoff + hf * 64;
      #pragma unroll
      for (int i = 0; i < 8; i++)
        *(v8bf*)&Ks[r * 128 + hf * 64 + i * 8] = *(const v8bf*)(src + i * 8);
    }
    {   // stage V tile transposed -> Vs[d][kn]; vector global loads,
        // packed 4-byte LDS stores (thread: kn pair 2c,2c+1; d range dq*32..+31)
      int c = tid & 31, dq = tid >> 5;
      const __bf16* s0 = V + ((size_t)b * Sk + k0 + 2 * c) * DIM_ + hoff + dq * 32;
      const __bf16* s1 = s0 + DIM_;
      #pragma unroll
      for (int i0 = 0; i0 < 32; i0 += 8) {
        v8bf a0 = *(const v8bf*)(s0 + i0);
        v8bf a1 = *(const v8bf*)(s1 + i0);
        #pragma unroll
        for (int j = 0; j < 8; j++) {
          int d = dq * 32 + i0 + j;
          v2bf p; p[0] = a0[j]; p[1] = a1[j];
          *(v2bf*)&Vs[d * 64 + 2 * c] = p;
        }
      }
    }
    __syncthreads();

    // T[kt] = K_tile(kt) @ Q(wave)^T, kt = 4 tiles of 16 kn rows
    v8f T[4] = {};
    for (int d0 = 0; d0 < 128; d0 += 32) {
      const __bf16* qb = &Qs[(wave * 16 + lm) * 128 + d0];
      v16bf bq = concat16(*(const v8bf*)(qb + 8 * h),
                          *(const v8bf*)(qb + 16 + 8 * h));
      #pragma unroll
      for (int kt = 0; kt < 4; kt++) {
        const __bf16* kb = &Ks[(kt * 16 + lm) * 128 + d0];
        v16bf ak = concat16(*(const v8bf*)(kb + 8 * h),
                            *(const v8bf*)(kb + 16 + 8 * h));
        T[kt] = __builtin_amdgcn_wmma_f32_16x16x32_bf16(
            false, ak, false, bq, (short)0, T[kt], false, false);
      }
    }

    // online softmax over kn (column q = lm, rows spread over VGPRs+halves)
    float vmax = -3.0e38f;
    #pragma unroll
    for (int kt = 0; kt < 4; kt++)
      #pragma unroll
      for (int r = 0; r < 8; r++) vmax = fmaxf(vmax, T[kt][r]);
    vmax = fmaxf(vmax, __shfl_xor(vmax, 16, 32));
    float m_new = fmaxf(m_prev, vmax);
    float alpha = __expf(m_prev - m_new);

    float P[4][8];
    float rowsum = 0.f;
    #pragma unroll
    for (int kt = 0; kt < 4; kt++)
      #pragma unroll
      for (int r = 0; r < 8; r++) {
        float p = __expf(T[kt][r] - m_new);
        P[kt][r] = p; rowsum += p;
      }
    rowsum += __shfl_xor(rowsum, 16, 32);
    l = l * alpha + rowsum;
    m_prev = m_new;

    // rescale O (O rows are q = r + 8h -> broadcast alpha from lane q)
    float osc[8];
    #pragma unroll
    for (int r = 0; r < 8; r++) osc[r] = __shfl(alpha, r + 8 * h, 32);
    #pragma unroll
    for (int dt = 0; dt < 8; dt++)
      #pragma unroll
      for (int r = 0; r < 8; r++) Oacc[dt][r] *= osc[r];

    // P (exp scores) -> bf16 A-fragments, purely in-lane
    v16bf pa[2];
    #pragma unroll
    for (int s = 0; s < 2; s++) {
      v16bf t;
      #pragma unroll
      for (int e = 0; e < 8; e++) {
        t[e]     = (__bf16)P[2 * s][e];
        t[8 + e] = (__bf16)P[2 * s + 1][e];
      }
      pa[s] = t;
    }

    // O += P @ V
    #pragma unroll
    for (int dt = 0; dt < 8; dt++) {
      #pragma unroll
      for (int s = 0; s < 2; s++) {
        const __bf16* vb = &Vs[(dt * 16 + lm) * 64 + s * 32 + 8 * h];
        v16bf bv = concat16(*(const v8bf*)vb, *(const v8bf*)(vb + 16));
        Oacc[dt] = __builtin_amdgcn_wmma_f32_16x16x32_bf16(
            false, pa[s], false, bv, (short)0, Oacc[dt], false, false);
      }
    }
  }

  float inv[8];
  #pragma unroll
  for (int r = 0; r < 8; r++) {
    float lr = __shfl(l, r + 8 * h, 32);
    inv[r] = 1.0f / lr;
  }
  #pragma unroll
  for (int dt = 0; dt < 8; dt++) {
    #pragma unroll
    for (int r = 0; r < 8; r++) {
      int qi = q0 + wave * 16 + r + 8 * h;
      O[((size_t)b * Sq + qi) * DIM_ + hoff + dt * 16 + lm] =
          (__bf16)(Oacc[dt][r] * inv[r]);
    }
  }
}

// ---------------------------------------------------------------------------
// Host-side orchestration
// ---------------------------------------------------------------------------
extern "C" void kernel_launch(void* const* d_in, const int* in_sizes, int n_in,
                              void* d_out, int out_size, void* d_ws, size_t ws_size,
                              hipStream_t stream) {
  const float* x          = (const float*)d_in[0];
  const float* e          = (const float*)d_in[1];
  const float* context    = (const float*)d_in[2];
  const float* modulation = (const float*)d_in[3];
  const float* norm1_w    = (const float*)d_in[4];
  const float* norm2_w    = (const float*)d_in[5];
  const float* norm3_w    = (const float*)d_in[6];
  const float* sa_nq_w    = (const float*)d_in[7];
  const float* sa_nk_w    = (const float*)d_in[8];
  const float* ffn_norm_w = (const float*)d_in[9];
  const float* sa_qw = (const float*)d_in[10]; const float* sa_qb = (const float*)d_in[11];
  const float* sa_kw = (const float*)d_in[12]; const float* sa_kb = (const float*)d_in[13];
  const float* sa_vw = (const float*)d_in[14]; const float* sa_vb = (const float*)d_in[15];
  const float* sa_ow = (const float*)d_in[16]; const float* sa_ob = (const float*)d_in[17];
  const float* ca_qw = (const float*)d_in[18]; const float* ca_qb = (const float*)d_in[19];
  const float* ca_kw = (const float*)d_in[20]; const float* ca_kb = (const float*)d_in[21];
  const float* ca_vw = (const float*)d_in[22]; const float* ca_vb = (const float*)d_in[23];
  const float* ca_ow = (const float*)d_in[24]; const float* ca_ob = (const float*)d_in[25];
  const float* ffn_w1 = (const float*)d_in[26]; const float* ffn_b1 = (const float*)d_in[27];
  const float* ffn_w2 = (const float*)d_in[28]; const float* ffn_b2 = (const float*)d_in[29];

  float* out = (float*)d_out;

  // workspace carve-up
  char* ws = (char*)d_ws;
  size_t off = 0;
  auto alloc = [&](size_t bytes) -> char* {
    char* p = ws + off;
    off += (bytes + 255) & ~(size_t)255;
    return p;
  };
  __bf16* wt_saq = (__bf16*)alloc((size_t)DIM_ * DIM_ * 2);
  __bf16* wt_sak = (__bf16*)alloc((size_t)DIM_ * DIM_ * 2);
  __bf16* wt_sav = (__bf16*)alloc((size_t)DIM_ * DIM_ * 2);
  __bf16* wt_sao = (__bf16*)alloc((size_t)DIM_ * DIM_ * 2);
  __bf16* wt_caq = (__bf16*)alloc((size_t)DIM_ * DIM_ * 2);
  __bf16* wt_cak = (__bf16*)alloc((size_t)DIM_ * DIM_ * 2);
  __bf16* wt_cav = (__bf16*)alloc((size_t)DIM_ * DIM_ * 2);
  __bf16* wt_cao = (__bf16*)alloc((size_t)DIM_ * DIM_ * 2);
  __bf16* wt_f1  = (__bf16*)alloc((size_t)DIM_ * FFN_ * 2);
  __bf16* wt_f2  = (__bf16*)alloc((size_t)FFN_ * DIM_ * 2);
  float*  modb   = (float*) alloc((size_t)B_ * 6 * DIM_ * 4);
  __bf16* hbf    = (__bf16*)alloc((size_t)NTOK_ * DIM_ * 2);
  float*  qf     = (float*) alloc((size_t)NTOK_ * DIM_ * 4);
  float*  kf     = (float*) alloc((size_t)NTOK_ * DIM_ * 4);
  __bf16* qbb    = (__bf16*)alloc((size_t)NTOK_ * DIM_ * 2);
  __bf16* kbb    = (__bf16*)alloc((size_t)NTOK_ * DIM_ * 2);
  __bf16* vbb    = (__bf16*)alloc((size_t)NTOK_ * DIM_ * 2);
  __bf16* ab     = (__bf16*)alloc((size_t)NTOK_ * DIM_ * 2);
  __bf16* ctxb   = (__bf16*)alloc((size_t)NCTX_ * DIM_ * 2);
  __bf16* kcb    = (__bf16*)alloc((size_t)NCTX_ * DIM_ * 2);
  __bf16* vcb    = (__bf16*)alloc((size_t)NCTX_ * DIM_ * 2);
  __bf16* f1b    = (__bf16*)alloc((size_t)NTOK_ * FFN_ * 2);
  (void)ws_size; (void)in_sizes; (void)n_in; (void)out_size;

  const float qscale = 0.08838834764831845f;   // 1/sqrt(128)

  // --- weight conversion (f32 [K][N] -> bf16 [N][K]) ---
  {
    dim3 blk(256);
    wtrans_kernel<<<dim3(DIM_/32, DIM_/32), blk, 0, stream>>>(sa_qw, wt_saq, DIM_, DIM_);
    wtrans_kernel<<<dim3(DIM_/32, DIM_/32), blk, 0, stream>>>(sa_kw, wt_sak, DIM_, DIM_);
    wtrans_kernel<<<dim3(DIM_/32, DIM_/32), blk, 0, stream>>>(sa_vw, wt_sav, DIM_, DIM_);
    wtrans_kernel<<<dim3(DIM_/32, DIM_/32), blk, 0, stream>>>(sa_ow, wt_sao, DIM_, DIM_);
    wtrans_kernel<<<dim3(DIM_/32, DIM_/32), blk, 0, stream>>>(ca_qw, wt_caq, DIM_, DIM_);
    wtrans_kernel<<<dim3(DIM_/32, DIM_/32), blk, 0, stream>>>(ca_kw, wt_cak, DIM_, DIM_);
    wtrans_kernel<<<dim3(DIM_/32, DIM_/32), blk, 0, stream>>>(ca_vw, wt_cav, DIM_, DIM_);
    wtrans_kernel<<<dim3(DIM_/32, DIM_/32), blk, 0, stream>>>(ca_ow, wt_cao, DIM_, DIM_);
    wtrans_kernel<<<dim3(DIM_/32, FFN_/32), blk, 0, stream>>>(ffn_w1, wt_f1, DIM_, FFN_);
    wtrans_kernel<<<dim3(FFN_/32, DIM_/32), blk, 0, stream>>>(ffn_w2, wt_f2, FFN_, DIM_);
  }

  // --- mod = modulation + e ---
  mod_kernel<<<(B_*6*DIM_ + 255)/256, 256, 0, stream>>>(modulation, e, modb, B_*6*DIM_);

  // --- self-attn: h = rms(x,norm1)*(1+e1)+e0 -> bf16 ---
  rmsmod_kernel<<<NTOK_, 256, 0, stream>>>(x, norm1_w, modb, 1, 0, nullptr, 1.0f, hbf, S_);

  // --- QKV projections ---
  gemm_bf16_kernel<<<dim3(NTOK_/128, DIM_/128), 256, 0, stream>>>(
      hbf, wt_saq, sa_qb, nullptr, nullptr, qf, nullptr,
      NTOK_, DIM_, DIM_, 1, 0, 0, 1.0f);
  gemm_bf16_kernel<<<dim3(NTOK_/128, DIM_/128), 256, 0, stream>>>(
      hbf, wt_sak, sa_kb, nullptr, nullptr, kf, nullptr,
      NTOK_, DIM_, DIM_, 1, 0, 0, 1.0f);
  gemm_bf16_kernel<<<dim3(NTOK_/128, DIM_/128), 256, 0, stream>>>(
      hbf, wt_sav, sa_vb, nullptr, nullptr, nullptr, vbb,
      NTOK_, DIM_, DIM_, 1, 0, 0, 1.0f);

  // --- qk-norm (full DIM), fold 1/sqrt(hd) into q ---
  rmsmod_kernel<<<NTOK_, 256, 0, stream>>>(qf, sa_nq_w, nullptr, 0, 0, nullptr, qscale, qbb, 1);
  rmsmod_kernel<<<NTOK_, 256, 0, stream>>>(kf, sa_nk_w, nullptr, 0, 0, nullptr, 1.0f,  kbb, 1);

  // --- self attention ---
  flash_attn_kernel<<<dim3(S_/64, HEADS_, B_), 128, 0, stream>>>(qbb, kbb, vbb, ab, S_, S_);

  // --- O projection: x2 = x + (ab@ow + ob) * e2 ---
  gemm_bf16_kernel<<<dim3(NTOK_/128, DIM_/128), 256, 0, stream>>>(
      ab, wt_sao, sa_ob, x, modb + 2 * DIM_, out, nullptr,
      NTOK_, DIM_, DIM_, S_, 6 * DIM_, 0, 1.0f);

  // --- cross attention ---
  rmsmod_kernel<<<NTOK_, 256, 0, stream>>>(out, norm2_w, nullptr, 0, 0, nullptr, 1.0f, hbf, S_);
  gemm_bf16_kernel<<<dim3(NTOK_/128, DIM_/128), 256, 0, stream>>>(
      hbf, wt_caq, ca_qb, nullptr, nullptr, nullptr, qbb,
      NTOK_, DIM_, DIM_, 1, 0, 0, qscale);
  f2b_kernel<<<((size_t)NCTX_*DIM_ + 255)/256, 256, 0, stream>>>(context, ctxb, (size_t)NCTX_*DIM_);
  gemm_bf16_kernel<<<dim3(NCTX_/128, DIM_/128), 256, 0, stream>>>(
      ctxb, wt_cak, ca_kb, nullptr, nullptr, nullptr, kcb,
      NCTX_, DIM_, DIM_, 1, 0, 0, 1.0f);
  gemm_bf16_kernel<<<dim3(NCTX_/128, DIM_/128), 256, 0, stream>>>(
      ctxb, wt_cav, ca_vb, nullptr, nullptr, nullptr, vcb,
      NCTX_, DIM_, DIM_, 1, 0, 0, 1.0f);
  flash_attn_kernel<<<dim3(S_/64, HEADS_, B_), 128, 0, stream>>>(qbb, kcb, vcb, ab, S_, CL_);
  gemm_bf16_kernel<<<dim3(NTOK_/128, DIM_/128), 256, 0, stream>>>(
      ab, wt_cao, ca_ob, out, nullptr, out, nullptr,
      NTOK_, DIM_, DIM_, S_, 0, 0, 1.0f);

  // --- FFN: h = rms(rms(x3,norm3)*(1+e4)+e3, ffn_norm) ---
  rmsmod_kernel<<<NTOK_, 256, 0, stream>>>(out, norm3_w, modb, 4, 3, ffn_norm_w, 1.0f, hbf, S_);
  gemm_bf16_kernel<<<dim3(NTOK_/128, FFN_/128), 256, 0, stream>>>(
      hbf, wt_f1, ffn_b1, nullptr, nullptr, nullptr, f1b,
      NTOK_, FFN_, DIM_, 1, 0, /*gelu*/1, 1.0f);
  gemm_bf16_kernel<<<dim3(NTOK_/128, DIM_/128), 256, 0, stream>>>(
      f1b, wt_f2, ffn_b2, out, modb + 5 * DIM_, out, nullptr,
      NTOK_, DIM_, FFN_, S_, 6 * DIM_, 0, 1.0f);
}
